// MultiScaleD_82403242541660
// MI455X (gfx1250) — matrix operand order repaired and checked by reading, not compiled, not run on hardware
//
#include <hip/hip_runtime.h>

typedef __attribute__((ext_vector_type(16))) __bf16 v16bf;
typedef __attribute__((ext_vector_type(8)))  __bf16 v8bf;
typedef __attribute__((ext_vector_type(8)))  float  v8f;
typedef __attribute__((ext_vector_type(4)))  unsigned int u32x4;
typedef __attribute__((ext_vector_type(8)))  int i32x8;
typedef __attribute__((ext_vector_type(4)))  int i32x4;

#if __has_builtin(__builtin_amdgcn_tensor_load_to_lds) && __has_builtin(__builtin_amdgcn_s_wait_tensorcnt)
#define HAS_TDM 1
#endif

static __host__ __device__ inline int cdiv_i(int a, int b) { return (a + b - 1) / b; }

// ---------------------------------------------------------------------------
// NCHW f32 -> NHWC bf16 with channel offset into a concat buffer
// ---------------------------------------------------------------------------
__global__ void cvt_nhwc_kernel(const float* __restrict__ x, __bf16* __restrict__ y,
                                int B, int C, int HW, int chanOff, int Ctot) {
    long i = (long)blockIdx.x * blockDim.x + threadIdx.x;
    long tot = (long)B * C * HW;
    if (i >= tot) return;
    int hw = (int)(i % HW);
    long t = i / HW;
    int c = (int)(t % C);
    int n = (int)(t / C);
    y[((long)n * HW + hw) * Ctot + chanOff + c] = (__bf16)x[i];
}

// OIHW f32 -> [co][tap][ci] bf16  (tap = kh*KW+kw)
__global__ void cvt_weight_kernel(const float* __restrict__ x, __bf16* __restrict__ y,
                                  int Cout, int Cin, int KHW) {
    long i = (long)blockIdx.x * blockDim.x + threadIdx.x;
    long tot = (long)Cout * Cin * KHW;
    if (i >= tot) return;
    int ci  = (int)(i % Cin);
    long t  = i / Cin;
    int tap = (int)(t % KHW);
    int co  = (int)(t / KHW);
    y[i] = (__bf16)x[((long)co * Cin + ci) * KHW + tap];
}

// ---------------------------------------------------------------------------
// BatchNorm statistics: one block per channel, reduce over B*HW (NCHW input)
// ---------------------------------------------------------------------------
__global__ void bn_stats_kernel(const float* __restrict__ x, float* __restrict__ mean,
                                float* __restrict__ rstd, int B, int C, int HW) {
    int c = blockIdx.x;
    float s = 0.0f, s2 = 0.0f;
    for (int n = 0; n < B; ++n) {
        const float* xp = x + ((long)n * C + c) * HW;
        for (int hw = threadIdx.x; hw < HW; hw += blockDim.x) {
            float v = xp[hw];
            s += v; s2 += v * v;
        }
    }
    __shared__ float sh[256], sh2[256];
    sh[threadIdx.x] = s; sh2[threadIdx.x] = s2;
    __syncthreads();
    for (int d = 128; d > 0; d >>= 1) {
        if ((int)threadIdx.x < d) {
            sh[threadIdx.x]  += sh[threadIdx.x + d];
            sh2[threadIdx.x] += sh2[threadIdx.x + d];
        }
        __syncthreads();
    }
    if (threadIdx.x == 0) {
        float N = (float)B * (float)HW;
        float m = sh[0] / N;
        float v = sh2[0] / N - m * m;
        mean[c] = m;
        rstd[c] = rsqrtf(v + 1e-5f);
    }
}

// BN apply + leaky relu; optional f32 NCHW dst (in-place ok), optional bf16
// NHWC dst with channel offset into a concat buffer.
__global__ void bn_apply_kernel(const float* __restrict__ x, const float* __restrict__ mean,
                                const float* __restrict__ rstd,
                                float* fdst, __bf16* bdst,
                                int B, int C, int HW, int chanOff, int Ctot) {
    long i = (long)blockIdx.x * blockDim.x + threadIdx.x;
    long tot = (long)B * C * HW;
    if (i >= tot) return;
    int hw = (int)(i % HW);
    long t = i / HW;
    int c = (int)(t % C);
    int n = (int)(t / C);
    float v = (x[i] - mean[c]) * rstd[c];
    v = (v > 0.0f) ? v : 0.2f * v;
    if (fdst) fdst[i] = v;
    if (bdst) bdst[((long)n * HW + hw) * Ctot + chanOff + c] = (__bf16)v;
}

// ---------------------------------------------------------------------------
// TDM: stage a [rows<=16] x [Cin] bf16 weight tile (row stride Ktot elements)
// into LDS at byte offset ldsByteOff.  D# built per CDNA5 ISA 8.3/8.4.
// 6-arg builtin form: (g0, g1, g2, g3, g_extra, cpol).
// ---------------------------------------------------------------------------
__device__ inline void tdm_stage_A(const __bf16* gsrc, unsigned ldsByteOff,
                                   int rowRem, int Cin, int Ktot) {
#ifdef HAS_TDM
    unsigned long long ga = (unsigned long long)(const void*)gsrc;
    u32x4 g0;
    g0[0] = 1u;                                              // count=1, user mode
    g0[1] = ldsByteOff;                                      // lds_addr
    g0[2] = (unsigned)(ga & 0xFFFFFFFFu);                    // global_addr lo
    g0[3] = (unsigned)((ga >> 32) & 0x01FFFFFFu) | 0x80000000u; // addr hi | type=2
    i32x8 g1;
    g1[0] = 0x00010000;                                      // data_size=1 (2 bytes)
    g1[1] = (int)(((unsigned)Cin & 0xFFFFu) << 16);          // tensor_dim0 lo16
    g1[2] = (int)((((unsigned)Cin >> 16) & 0xFFFFu) |
                  (((unsigned)rowRem & 0xFFFFu) << 16));     // dim0 hi16 | dim1 lo16
    g1[3] = (int)(((unsigned)Cin & 0xFFFFu) << 16);          // dim1 hi16=0 | tile_dim0
    g1[4] = 16;                                              // tile_dim1=16, tile_dim2=0
    g1[5] = Ktot;                                            // tensor_dim0_stride lo32
    g1[6] = 0;                                               // stride hi16 | dim1_stride lo16
    g1[7] = 0;
    i32x4 gz4 = {};
    i32x8 gz8 = {};
    __builtin_amdgcn_tensor_load_to_lds(g0, g1, gz4, gz4, gz8, 0);
#else
    (void)gsrc; (void)ldsByteOff; (void)rowRem; (void)Cin; (void)Ktot;
#endif
}

static __device__ inline v16bf cat16bf(v8bf lo, v8bf hi) {
    v16bf r;
#pragma unroll
    for (int i = 0; i < 8; ++i) { r[i] = lo[i]; r[8 + i] = hi[i]; }
    return r;
}

// ---------------------------------------------------------------------------
// Implicit-GEMM conv via V_WMMA_F32_16X16X32_BF16.
// K-order: k = tap*Cin + ci (weights pre-packed [co][tap][ci]); NHWC bf16
// activations.  Each wave computes 16 couts x 32 positions (two C tiles) so
// every LDS A-fragment feeds TWO WMMAs.  A slab staged per tap by the Tensor
// Data Mover, double-buffered with s_wait_tensorcnt; 8 waves share the slab.
// ---------------------------------------------------------------------------
__global__ void conv_wmma_kernel(const __bf16* __restrict__ in, int inChanOff, int inCtot,
                                 const __bf16* __restrict__ w, float* __restrict__ out,
                                 int B, int Cin, int H, int W,
                                 int Cout, int KH, int KW, int stride, int pad,
                                 int Ho, int Wo, long outBatchStride, long outChanStride) {
    extern __shared__ __bf16 smem[];
    const int lane = threadIdx.x & 31;
    const int wv   = threadIdx.x >> 5;
    const int posTile = blockIdx.x * (blockDim.x >> 5) + wv;   // 32 positions per wave
    const int coTile  = blockIdx.y;
    const int npos = B * Ho * Wo;
    const int KHW  = KH * KW;
    const int Ktot = Cin * KHW;
    const int half = lane >> 4;
    const int l15  = lane & 15;
    const int slabElems = 16 * Cin;

    // two output positions per lane (N columns of the two C tiles)
    int p0 = posTile * 32 + l15;
    int p1 = p0 + 16;
    bool pv0 = p0 < npos, pv1 = p1 < npos;
    int pc0 = pv0 ? p0 : 0, pc1 = pv1 ? p1 : 0;
    int n0 = pc0 / (Ho * Wo), r0 = pc0 % (Ho * Wo);
    int n1 = pc1 / (Ho * Wo), r1 = pc1 % (Ho * Wo);
    int ho0 = r0 / Wo, wo0 = r0 % Wo;
    int ho1 = r1 / Wo, wo1 = r1 % Wo;
    int yb0 = ho0 * stride - pad, xb0 = wo0 * stride - pad;
    int yb1 = ho1 * stride - pad, xb1 = wo1 * stride - pad;

    int rowRem = Cout - coTile * 16;
    if (rowRem > 16) rowRem = 16;
    const __bf16* wbase = w + (long)coTile * 16 * Ktot;
    const int koffA = half * 16;    // lanes 0-15: ci 0..15; 16-31: ci 16..31
    const int koffB = half * 16;    // lanes 0-15: ci 0..15; 16-31: ci 16..31

    unsigned ldsBase = (unsigned)(unsigned long long)(void*)smem;

#ifdef HAS_TDM
    if (wv == 0) tdm_stage_A(wbase, ldsBase, rowRem, Cin, Ktot);   // prologue: tap 0 -> buf 0
#endif

    v8f acc0 = {}, acc1 = {};
    const v8bf bz = {};
    for (int tap = 0; tap < KHW; ++tap) {
        int buf = tap & 1;
#ifdef HAS_TDM
        if (wv == 0) {
            if (tap + 1 < KHW) {
                tdm_stage_A(wbase + (long)(tap + 1) * Cin,
                            ldsBase + (unsigned)((buf ^ 1) * slabElems * 2),
                            rowRem, Cin, Ktot);
                __builtin_amdgcn_s_wait_tensorcnt(1);   // tap's slab has landed
            } else {
                __builtin_amdgcn_s_wait_tensorcnt(0);
            }
        }
        __syncthreads();
        const __bf16* slab = smem + buf * slabElems;
#else
        __syncthreads();
        for (int t = threadIdx.x; t < slabElems; t += blockDim.x) {
            int r2 = t / Cin, cc = t % Cin;
            smem[t] = (r2 < rowRem) ? wbase[(long)r2 * Ktot + (long)tap * Cin + cc]
                                    : (__bf16)0.0f;
        }
        __syncthreads();
        const __bf16* slab = smem;
        (void)buf;
#endif
        int kh = tap / KW, kw = tap - (tap / KW) * KW;
        int y0 = yb0 + kh, x0 = xb0 + kw;
        int y1 = yb1 + kh, x1 = xb1 + kw;
        bool ok0 = pv0 && (y0 >= 0) && (y0 < H) && (x0 >= 0) && (x0 < W);
        bool ok1 = pv1 && (y1 >= 0) && (y1 < H) && (x1 >= 0) && (x1 < W);
        int yy0 = ok0 ? y0 : 0, xx0 = ok0 ? x0 : 0;
        int yy1 = ok1 ? y1 : 0, xx1 = ok1 ? x1 : 0;
        const v8bf* bp0 = (const v8bf*)(in + (((long)n0 * H + yy0) * W + xx0) * inCtot
                                           + inChanOff + koffB);
        const v8bf* bp1 = (const v8bf*)(in + (((long)n1 * H + yy1) * W + xx1) * inCtot
                                           + inChanOff + koffB);
        const v8bf* ap = (const v8bf*)(slab + l15 * Cin + koffA);
#pragma unroll 4
        for (int cb = 0; cb < Cin; cb += 32) {
            __builtin_prefetch((const __bf16*)bp0 + 32, 0, 3);    // global_prefetch_b8
            v16bf afrag = cat16bf(ap[0], ap[1]);                  // 2x ds_load_b128
            v8bf b0lo = ok0 ? bp0[0] : bz;                        // global_load_b128 x2
            v8bf b0hi = ok0 ? bp0[1] : bz;
            v8bf b1lo = ok1 ? bp1[0] : bz;                        // global_load_b128 x2
            v8bf b1hi = ok1 ? bp1[1] : bz;
            acc0 = __builtin_amdgcn_wmma_f32_16x16x32_bf16(
                       false, afrag, false, cat16bf(b0lo, b0hi), (short)0, acc0, false, false);
            acc1 = __builtin_amdgcn_wmma_f32_16x16x32_bf16(
                       false, afrag, false, cat16bf(b1lo, b1hi), (short)0, acc1, false, false);
            ap += 4;                     // +32 ci
            bp0 += 4; bp1 += 4;
        }
        __syncthreads();                 // slab free before TDM overwrites it
    }

    // C/D layout: VGPR r -> M = r + 8*half, N = lane&15
    int mbase = half * 8;
#pragma unroll
    for (int r2 = 0; r2 < 8; ++r2) {
        int coS = coTile * 16 + mbase + r2;
        if (coS < Cout) {
            if (pv0) out[(long)n0 * outBatchStride + (long)coS * outChanStride
                         + (long)ho0 * Wo + wo0] = acc0[r2];
            if (pv1) out[(long)n1 * outBatchStride + (long)coS * outChanStride
                         + (long)ho1 * Wo + wo1] = acc1[r2];
        }
    }
}

// ---------------------------------------------------------------------------
// ROI-align (8x8 bins, 2x2 samples) fused with segment-sum accumulation.
// ---------------------------------------------------------------------------
__global__ void roi_accum_kernel(const float* __restrict__ feat, const float* __restrict__ boxes,
                                 float* __restrict__ psum, float* __restrict__ pcnt,
                                 int C, int H, int W) {
    int k = blockIdx.x;
    __shared__ int   sx0[16], sx1[16], sy0[16], sy1[16];
    __shared__ float slx[16], sly[16];
    __shared__ int   sbidx;
    const float* bx = boxes + (long)k * 5;
    if (threadIdx.x == 0) sbidx = (int)bx[0];
    if (threadIdx.x < 16) {
        int s = threadIdx.x;
        float x1 = bx[1], y1 = bx[2], x2 = bx[3], y2 = bx[4];
        float rw = fmaxf(x2 - x1, 1.0f), rh = fmaxf(y2 - y1, 1.0f);
        float g = ((float)s + 0.5f) * 0.5f;
        float xs = fminf(fmaxf(x1 + g * (rw * 0.125f), 0.0f), (float)(W - 1));
        float ys = fminf(fmaxf(y1 + g * (rh * 0.125f), 0.0f), (float)(H - 1));
        int x0 = (int)floorf(xs), y0 = (int)floorf(ys);
        sx0[s] = x0; sy0[s] = y0;
        sx1[s] = (x0 + 1 < W - 1) ? x0 + 1 : W - 1;
        sy1[s] = (y0 + 1 < H - 1) ? y0 + 1 : H - 1;
        slx[s] = xs - (float)x0;
        sly[s] = ys - (float)y0;
    }
    __syncthreads();
    int bidx = sbidx;
    const float* fb = feat + (long)bidx * C * H * W;
    for (int o = threadIdx.x; o < C * 64; o += blockDim.x) {
        int c = o >> 6, bin = o & 63, by = bin >> 3, bxn = bin & 7;
        const float* fc = fb + (long)c * H * W;
        float sum = 0.0f;
#pragma unroll
        for (int sy = 0; sy < 2; ++sy) {
            int gy = by * 2 + sy;
            int y0 = sy0[gy], y1i = sy1[gy];
            float ly = sly[gy];
#pragma unroll
            for (int sx = 0; sx < 2; ++sx) {
                int gx = bxn * 2 + sx;
                int x0 = sx0[gx], x1i = sx1[gx];
                float lx = slx[gx];
                float v00 = fc[y0 * W + x0],  v01 = fc[y0 * W + x1i];
                float v10 = fc[y1i * W + x0], v11 = fc[y1i * W + x1i];
                sum += (1.0f - ly) * (1.0f - lx) * v00 + (1.0f - ly) * lx * v01 +
                       ly * (1.0f - lx) * v10 + ly * lx * v11;
            }
        }
        atomicAdd(&psum[((long)bidx * C + c) * 64 + bin], sum * 0.25f);
    }
    if (threadIdx.x == 0) atomicAdd(&pcnt[bidx], 1.0f);
}

__global__ void zero_f32_kernel(float* p, long n) {
    long i = (long)blockIdx.x * blockDim.x + threadIdx.x;
    if (i < n) p[i] = 0.0f;
}

// pooled = psum / max(cnt,1); psum is [32][128][64] NCHW -> NHWC bf16 [32][64][128]
__global__ void pool_final_kernel(const float* __restrict__ psum, const float* __restrict__ pcnt,
                                  __bf16* __restrict__ out) {
    int i = blockIdx.x * blockDim.x + threadIdx.x;
    if (i >= 262144) return;
    int n = i >> 13, c = (i >> 6) & 127, bin = i & 63;
    out[((long)(n << 6) + bin) * 128 + c] = (__bf16)(psum[i] / fmaxf(pcnt[n], 1.0f));
}

// ---------------------------------------------------------------------------
static void launch_conv(hipStream_t s, const __bf16* in, int chanOff, int Ctot,
                        const __bf16* w, float* out,
                        int B, int Cin, int H, int W,
                        int Cout, int KH, int KW, int st, int pad, int Ho, int Wo,
                        long outBS, long outCS) {
    int npos = B * Ho * Wo;
    int posTiles = cdiv_i(npos, 32);                          // 32 positions per wave
    dim3 grid(cdiv_i(posTiles, 8), cdiv_i(Cout, 16));
    size_t shBytes = (size_t)2 * 16 * Cin * sizeof(__bf16);   // double-buffered A slab
    conv_wmma_kernel<<<grid, 256, shBytes, s>>>(in, chanOff, Ctot, w, out,
                                                B, Cin, H, W, Cout, KH, KW, st, pad,
                                                Ho, Wo, outBS, outCS);
}

extern "C" void kernel_launch(void* const* d_in, const int* in_sizes, int n_in,
                              void* d_out, int out_size, void* d_ws, size_t ws_size,
                              hipStream_t stream) {
    (void)n_in; (void)out_size; (void)ws_size;
    const float* feat0  = (const float*)d_in[0];   // [32][64][64][64]
    const float* feat1  = (const float*)d_in[1];   // [32][128][32][32]
    const float* bbox_s = (const float*)d_in[2];
    const float* bbox_m = (const float*)d_in[3];
    const float* bbox_l = (const float*)d_in[4];
    float* out = (float*)d_out;

    size_t off = 0;
    auto alloc = [&](size_t bytes) -> void* {
        void* p = (char*)d_ws + off;
        off += (bytes + 255) & ~(size_t)255;
        return p;
    };

    __bf16* feat0_nh = (__bf16*)alloc((size_t)8388608 * 2);   // NHWC [32][4096][64]
    __bf16* cat32    = (__bf16*)alloc((size_t)8388608 * 2);   // NHWC [32][1024][256]
    __bf16* cat16    = (__bf16*)alloc((size_t)4194304 * 2);   // NHWC [32][256][512]
    float*  x1raw    = (float*)alloc((size_t)4194304 * 4);    // NCHW [32][128][32][32]
    float*  x2raw    = (float*)alloc((size_t)2097152 * 4);    // [32][256][16][16]
    float*  x3raw    = (float*)alloc((size_t)1048576 * 4);    // [32][512][8][8]
    __bf16* x3bf     = (__bf16*)alloc((size_t)1048576 * 2);   // NHWC [32][64][512]
    float*  y1raw    = (float*)alloc((size_t)2097152 * 4);
    float*  y2raw    = (float*)alloc((size_t)1048576 * 4);
    __bf16* y2bf     = (__bf16*)alloc((size_t)1048576 * 2);
    float*  a64raw   = (float*)alloc((size_t)16777216 * 4);   // [32][128][64][64]
    float*  a32raw   = (float*)alloc((size_t)4194304 * 4);
    float*  a16raw   = (float*)alloc((size_t)1048576 * 4);
    float*  psum     = (float*)alloc((size_t)(262144 + 32) * 4);
    float*  pcnt     = psum + 262144;
    __bf16* pooledbf = (__bf16*)alloc((size_t)262144 * 2);    // NHWC [32][64][128]

    // weights -> bf16 [co][tap][ci]
    const int wco[11]  = {128, 256, 512, 1, 256, 512, 1, 128, 128, 128, 1};
    const int wci[11]  = {64, 128, 256, 512, 128, 256, 512, 64, 256, 512, 128};
    const int wkhw[11] = {16, 16, 16, 16, 16, 16, 16, 9, 9, 9, 16};
    __bf16* wbf[11];
    for (int i = 0; i < 11; ++i) {
        int nel = in_sizes[5 + i];
        wbf[i] = (__bf16*)alloc((size_t)nel * 2);
        cvt_weight_kernel<<<cdiv_i(nel, 256), 256, 0, stream>>>(
            (const float*)d_in[5 + i], wbf[i], wco[i], wci[i], wkhw[i]);
    }

    float* st_x1  = (float*)alloc(2 * 128 * 4);
    float* st_x2  = (float*)alloc(2 * 256 * 4);
    float* st_x3  = (float*)alloc(2 * 512 * 4);
    float* st_y1  = (float*)alloc(2 * 256 * 4);
    float* st_y2  = (float*)alloc(2 * 512 * 4);
    float* st_a64 = (float*)alloc(2 * 128 * 4);
    float* st_a32 = (float*)alloc(2 * 128 * 4);
    float* st_a16 = (float*)alloc(2 * 128 * 4);

    // ---- input conversions (NCHW f32 -> NHWC bf16) -------------------------
    cvt_nhwc_kernel<<<cdiv_i(8388608, 256), 256, 0, stream>>>(feat0, feat0_nh, 32, 64, 4096, 0, 64);
    cvt_nhwc_kernel<<<cdiv_i(4194304, 256), 256, 0, stream>>>(feat1, cat32, 32, 128, 1024, 128, 256);

    // ---- discriminator 0 chain --------------------------------------------
    launch_conv(stream, feat0_nh, 0, 64, wbf[0], x1raw, 32, 64, 64, 64,
                128, 4, 4, 2, 1, 32, 32, 131072, 1024);
    bn_stats_kernel<<<128, 256, 0, stream>>>(x1raw, st_x1, st_x1 + 128, 32, 128, 1024);
    bn_apply_kernel<<<cdiv_i(4194304, 256), 256, 0, stream>>>(x1raw, st_x1, st_x1 + 128,
        (float*)nullptr, cat32, 32, 128, 1024, 0, 256);

    launch_conv(stream, cat32, 0, 256, wbf[1], x2raw, 32, 128, 32, 32,
                256, 4, 4, 2, 1, 16, 16, 65536, 256);
    bn_stats_kernel<<<256, 256, 0, stream>>>(x2raw, st_x2, st_x2 + 256, 32, 256, 256);
    bn_apply_kernel<<<cdiv_i(2097152, 256), 256, 0, stream>>>(x2raw, st_x2, st_x2 + 256,
        (float*)nullptr, cat16, 32, 256, 256, 0, 512);

    launch_conv(stream, cat16, 0, 512, wbf[2], x3raw, 32, 256, 16, 16,
                512, 4, 4, 2, 1, 8, 8, 32768, 64);
    bn_stats_kernel<<<512, 256, 0, stream>>>(x3raw, st_x3, st_x3 + 512, 32, 512, 64);
    bn_apply_kernel<<<cdiv_i(1048576, 256), 256, 0, stream>>>(x3raw, st_x3, st_x3 + 512,
        (float*)nullptr, x3bf, 32, 512, 64, 0, 512);

    launch_conv(stream, x3bf, 0, 512, wbf[3], out, 32, 512, 8, 8,
                1, 4, 4, 1, 0, 5, 5, 50, 25);                 // logit0 -> out[n][0]

    // ---- discriminator 1 chain --------------------------------------------
    launch_conv(stream, cat32, 128, 256, wbf[4], y1raw, 32, 128, 32, 32,
                256, 4, 4, 2, 1, 16, 16, 65536, 256);
    bn_stats_kernel<<<256, 256, 0, stream>>>(y1raw, st_y1, st_y1 + 256, 32, 256, 256);
    bn_apply_kernel<<<cdiv_i(2097152, 256), 256, 0, stream>>>(y1raw, st_y1, st_y1 + 256,
        (float*)nullptr, cat16, 32, 256, 256, 256, 512);

    launch_conv(stream, cat16, 256, 512, wbf[5], y2raw, 32, 256, 16, 16,
                512, 4, 4, 2, 1, 8, 8, 32768, 64);
    bn_stats_kernel<<<512, 256, 0, stream>>>(y2raw, st_y2, st_y2 + 512, 32, 512, 64);
    bn_apply_kernel<<<cdiv_i(1048576, 256), 256, 0, stream>>>(y2raw, st_y2, st_y2 + 512,
        (float*)nullptr, y2bf, 32, 512, 64, 0, 512);

    launch_conv(stream, y2bf, 0, 512, wbf[6], out + 25, 32, 512, 8, 8,
                1, 4, 4, 1, 0, 5, 5, 50, 25);                 // logit1 -> out[n][1]

    // ---- lsb feature heads -------------------------------------------------
    launch_conv(stream, feat0_nh, 0, 64, wbf[7], a64raw, 32, 64, 64, 64,
                128, 3, 3, 1, 1, 64, 64, 524288, 4096);
    bn_stats_kernel<<<128, 256, 0, stream>>>(a64raw, st_a64, st_a64 + 128, 32, 128, 4096);
    bn_apply_kernel<<<cdiv_i(16777216, 256), 256, 0, stream>>>(a64raw, st_a64, st_a64 + 128,
        a64raw, (__bf16*)nullptr, 32, 128, 4096, 0, 128);

    launch_conv(stream, cat32, 0, 256, wbf[8], a32raw, 32, 256, 32, 32,
                128, 3, 3, 1, 1, 32, 32, 131072, 1024);
    bn_stats_kernel<<<128, 256, 0, stream>>>(a32raw, st_a32, st_a32 + 128, 32, 128, 1024);
    bn_apply_kernel<<<cdiv_i(4194304, 256), 256, 0, stream>>>(a32raw, st_a32, st_a32 + 128,
        a32raw, (__bf16*)nullptr, 32, 128, 1024, 0, 128);

    launch_conv(stream, cat16, 0, 512, wbf[9], a16raw, 32, 512, 16, 16,
                128, 3, 3, 1, 1, 16, 16, 32768, 256);
    bn_stats_kernel<<<128, 256, 0, stream>>>(a16raw, st_a16, st_a16 + 128, 32, 128, 256);
    bn_apply_kernel<<<cdiv_i(1048576, 256), 256, 0, stream>>>(a16raw, st_a16, st_a16 + 128,
        a16raw, (__bf16*)nullptr, 32, 128, 256, 0, 128);

    // ---- ROI align + fused segment mean ------------------------------------
    zero_f32_kernel<<<cdiv_i(262144 + 32, 256), 256, 0, stream>>>(psum, 262144 + 32);
    roi_accum_kernel<<<1024, 256, 0, stream>>>(a64raw, bbox_s, psum, pcnt, 128, 64, 64);
    roi_accum_kernel<<<1024, 256, 0, stream>>>(a32raw, bbox_m, psum, pcnt, 128, 32, 32);
    roi_accum_kernel<<<1024, 256, 0, stream>>>(a16raw, bbox_l, psum, pcnt, 128, 16, 16);
    pool_final_kernel<<<cdiv_i(262144, 256), 256, 0, stream>>>(psum, pcnt, pooledbf);

    // final conv: 128->1, 4x4 s1 p0, 8x8 -> 5x5, out offset 1600
    launch_conv(stream, pooledbf, 0, 128, wbf[10], out + 1600, 32, 128, 8, 8,
                1, 4, 4, 1, 0, 5, 5, 25, 25);
}
